// NeuralMemory_12567074308567
// MI455X (gfx1250) — matrix operand order repaired and checked by reading, compiled, not run on hardware
//
#include <hip/hip_runtime.h>

// ---------------- problem constants ----------------
#define BB   2
#define LL   4096
#define DIMD 512
#define HH   4
#define DHD  128
#define CC   64
#define NN   64          // chunks per sequence
#define BHH  8           // B*H
#define MM   8192        // B*L
#define EPSF 1e-6f

typedef __attribute__((ext_vector_type(16))) __bf16 v16bf;
typedef __attribute__((ext_vector_type(8)))  float  v8f;

__device__ inline float sigmoidf_(float x) { return 1.0f / (1.0f + __expf(-x)); }
__device__ inline float siluf_(float x)    { return x * sigmoidf_(x); }

__device__ inline v8f wmma_bf16_(v16bf a, v16bf b, v8f c) {
  return __builtin_amdgcn_wmma_f32_16x16x32_bf16(false, a, false, b, (short)0, c, false, false);
}

// ---------------- WMMA fragment loaders (wave32) ----------------
// A-matrix 16x32 bf16 layout (ISA 7.12.2): lane = M + 16*kg; per lane the 16
// elements are K = 8*kg + [0..7] then K = 8*kg + 16 + [0..7]  (contiguous runs
// -> compiler emits two ds_load_b128 per fragment).
__device__ inline v16bf frag_a_rm(const __bf16* A, int ldA, int m_base, int k_base, int lane) {
  int m = m_base + (lane & 15), kg = lane >> 4;
  const __bf16* p = A + (size_t)m * ldA + k_base + kg * 8;
  v16bf f;
#pragma unroll
  for (int e = 0; e < 8; ++e) f[e] = p[e];
#pragma unroll
  for (int e = 0; e < 8; ++e) f[e + 8] = p[e + 16];
  return f;
}
// A operand where logical A[m][k] = T[k][m] (free transpose, strided gather)
__device__ inline v16bf frag_a_tr(const __bf16* T, int ldT, int m_base, int k_base, int lane) {
  int m = m_base + (lane & 15), kg = lane >> 4;
  v16bf f;
#pragma unroll
  for (int e = 0; e < 16; ++e) {
    int K = k_base + ((e < 8) ? e : e + 8) + 8 * kg;
    f[e] = T[(size_t)K * ldT + m];
  }
  return f;
}
// B-matrix 32x16 bf16: lane = N + 16*kg; per lane K = 8*kg + [0..7] then K = 8*kg + 16 + [0..7].
// Natural row-major B[k][n] (strided gather):
__device__ inline v16bf frag_b_rm(const __bf16* Bm, int ldB, int n_base, int k_base, int lane) {
  int n = n_base + (lane & 15), kg = lane >> 4;
  v16bf f;
#pragma unroll
  for (int e = 0; e < 16; ++e) {
    int K = k_base + ((e < 8) ? e : e + 8) + 8 * kg;
    f[e] = Bm[(size_t)K * ldB + n];
  }
  return f;
}
// B stored as Bt[n][k] (contiguous loads; also realizes B = Wᵀ from row-major W)
__device__ inline v16bf frag_b_tr(const __bf16* Bt, int ldBt, int n_base, int k_base, int lane) {
  int n = n_base + (lane & 15), kg = lane >> 4;
  const __bf16* p = Bt + (size_t)n * ldBt + k_base + kg * 8;
  v16bf f;
#pragma unroll
  for (int e = 0; e < 16; ++e) f[e] = p[(e < 8) ? e : e + 8];
  return f;
}

// ============================================================
// 1) rmsnorm for store & retrieve paths; emits bf16 s and shifted r
// ============================================================
__global__ __launch_bounds__(256) void rmsnorm_k(const float* __restrict__ seq,
                                                 const float* __restrict__ ssc,
                                                 const float* __restrict__ rsc,
                                                 __bf16* __restrict__ s_bf,
                                                 __bf16* __restrict__ rsh_bf) {
  int tok = blockIdx.x;                 // b*L + l
  int b = tok >> 12, l = tok & (LL - 1);
  int tid = threadIdx.x;
  __shared__ float red[256];
  const float* x = seq + (size_t)tok * DIMD;
  float p = x[tid] * x[tid] + x[tid + 256] * x[tid + 256];
  red[tid] = p;
  __syncthreads();
  for (int s2 = 128; s2 > 0; s2 >>= 1) {
    if (tid < s2) red[tid] += red[tid + s2];
    __syncthreads();
  }
  float rinv = rsqrtf(red[0] * (1.0f / DIMD) + EPSF);
#pragma unroll
  for (int rep = 0; rep < 2; ++rep) {
    int d = tid + rep * 256;
    float xv = x[d] * rinv;
    s_bf[(size_t)tok * DIMD + d] = (__bf16)(xv * ssc[d]);
    float rv = xv * rsc[d];
    if (l >= CC - 1)
      rsh_bf[((size_t)b * LL + (l - (CC - 1))) * DIMD + d] = (__bf16)rv;
  }
  if (l < CC - 1) {  // zero the padded tail rows of r_shift
    size_t rb = ((size_t)b * LL + (LL - (CC - 1) + l)) * DIMD;
    rsh_bf[rb + tid] = (__bf16)0.0f;
    rsh_bf[rb + tid + 256] = (__bf16)0.0f;
  }
}

// ============================================================
// 2) fp32 -> bf16 weight conversion
// ============================================================
__global__ void cvt_bf16_k(const float* __restrict__ src, __bf16* __restrict__ dst, int n) {
  int i = blockIdx.x * 256 + threadIdx.x;
  if (i < n) dst[i] = (__bf16)src[i];
}

// ============================================================
// 3) per-token tiny projections: adaptive_lr (from s) and gate (from r_shift)
// ============================================================
__global__ __launch_bounds__(64) void tokproj_k(const __bf16* __restrict__ s_bf,
                                                const __bf16* __restrict__ rsh_bf,
                                                const float* __restrict__ w_ada,
                                                const float* __restrict__ w_gate,
                                                float* __restrict__ lr_g,
                                                float* __restrict__ gate_g) {
  int tok = blockIdx.x;
  int b = tok >> 12, l = tok & (LL - 1);
  int tid = threadIdx.x;
  __shared__ float red[8][64];
  float p[8];
#pragma unroll
  for (int o = 0; o < 8; ++o) p[o] = 0.0f;
#pragma unroll
  for (int jj = 0; jj < 8; ++jj) {
    int d = tid * 8 + jj;
    float sv = (float)s_bf[(size_t)tok * DIMD + d];
    float rv = (float)rsh_bf[(size_t)tok * DIMD + d];
#pragma unroll
    for (int h = 0; h < HH; ++h) {
      p[h]      += sv * w_ada[d * HH + h];
      p[4 + h]  += rv * w_gate[d * HH + h];
    }
  }
#pragma unroll
  for (int o = 0; o < 8; ++o) red[o][tid] = p[o];
  __syncthreads();
  if (tid < 8) {
    float s = 0.0f;
    for (int i = 0; i < 64; ++i) s += red[tid][i];
    float sg = sigmoidf_(s);
    if (tid < 4) lr_g[((size_t)(b * HH + tid)) * LL + l] = sg * 0.01f;   // MAX_LR
    else         gate_g[((size_t)(b * HH + (tid - 4))) * LL + l] = sg;
  }
}

// ============================================================
// 4) per-chunk mean of s + ada_mom / decay projections
// ============================================================
__global__ __launch_bounds__(256) void chunkstat_k(const __bf16* __restrict__ s_bf,
                                                   const float* __restrict__ w_mom,
                                                   const float* __restrict__ w_dec,
                                                   float* __restrict__ adam,
                                                   float* __restrict__ dec) {
  int bn = blockIdx.x;                 // b*N + n
  int b = bn >> 6, n = bn & (NN - 1);
  int tid = threadIdx.x;
  __shared__ float mean[DIMD];
  __shared__ float red2[8][256];
#pragma unroll
  for (int rep = 0; rep < 2; ++rep) {
    int d = tid + rep * 256;
    float acc = 0.0f;
    for (int t = 0; t < CC; ++t)
      acc += (float)s_bf[((size_t)b * LL + n * CC + t) * DIMD + d];
    mean[d] = acc * (1.0f / CC);
  }
  __syncthreads();
  float p[8];
#pragma unroll
  for (int o = 0; o < 8; ++o) p[o] = 0.0f;
#pragma unroll
  for (int rep = 0; rep < 2; ++rep) {
    int d = tid + rep * 256;
#pragma unroll
    for (int h = 0; h < HH; ++h) {
      p[h]     += mean[d] * w_mom[d * HH + h];
      p[4 + h] += mean[d] * w_dec[d * HH + h];
    }
  }
#pragma unroll
  for (int o = 0; o < 8; ++o) red2[o][tid] = p[o];
  __syncthreads();
  if (tid < 8) {
    float s = 0.0f;
    for (int i = 0; i < 256; ++i) s += red2[tid][i];
    float sg = sigmoidf_(s);
    int h = tid & 3;
    if (tid < 4) adam[((size_t)(b * HH + h)) * NN + n] = sg;
    else         dec [((size_t)(b * HH + h)) * NN + n] = sg;
  }
}

// ============================================================
// 5) big bf16 WMMA GEMM (M x K) @ (K x N), 128x128 block tile,
//    double-buffered LDS with register staging + GL2 prefetch.
//    mode 0: kv projection  -> keys(bf16, head-major) + vals(f32, head-major)
//    mode 1: q  projection  -> q(bf16, head-major)
//    mode 2: combine        -> d_out with +63 token shift
// ============================================================
__global__ __launch_bounds__(256) void gemm_proj_k(const __bf16* __restrict__ A,
                                                   const __bf16* __restrict__ Bm,
                                                   int N, int K, int mode,
                                                   __bf16* __restrict__ out_bf,
                                                   float* __restrict__ out_f) {
  __shared__ __bf16 lds_a[2][128][32];      // A tiles (row-major)
  __shared__ __bf16 lds_bt[2][128][32];     // B tiles transposed: [n][k]
  int m0 = blockIdx.x * 128, n0 = blockIdx.y * 128;
  int tid = threadIdx.x, lane = tid & 31, wv = tid >> 5;
  int m_off = (wv & 3) * 32, n_off = (wv >> 2) * 64;   // wave -> 32x64 sub-tile
  v8f acc[2][4] = {};

  // register staging (16 bf16 = one 32B run per thread for each tile)
  __bf16 ra[16], rb[16];
  int arow = tid >> 1, ac0 = (tid & 1) * 16;  // 128 rows x 32 cols
  int bkk  = tid >> 3, bnc0 = (tid & 7) * 16; // 32 k-rows x 128 n-cols

  {  // stage tile 0
    const __bf16* ga = A + (size_t)(m0 + arow) * K + ac0;
    const __bf16* gb = Bm + (size_t)bkk * N + n0 + bnc0;
#pragma unroll
    for (int j = 0; j < 16; ++j) ra[j] = ga[j];
#pragma unroll
    for (int j = 0; j < 16; ++j) rb[j] = gb[j];
#pragma unroll
    for (int j = 0; j < 16; ++j) lds_a[0][arow][ac0 + j] = ra[j];
#pragma unroll
    for (int j = 0; j < 16; ++j) lds_bt[0][bnc0 + j][bkk] = rb[j];
  }
  __syncthreads();

  int nsteps = K / 32;
  for (int it = 0; it < nsteps; ++it) {
    int cur = it & 1, nxt = cur ^ 1;
    if (it + 1 < nsteps) {       // issue next-tile global loads before compute
      int k0 = (it + 1) * 32;
      const __bf16* ga = A + (size_t)(m0 + arow) * K + k0 + ac0;
      const __bf16* gb = Bm + (size_t)(k0 + bkk) * N + n0 + bnc0;
#pragma unroll
      for (int j = 0; j < 16; ++j) ra[j] = ga[j];
#pragma unroll
      for (int j = 0; j < 16; ++j) rb[j] = gb[j];
      if (it + 2 < nsteps) {     // pull tile t+2 into GL2 (global_prefetch_b8)
        __builtin_prefetch(A + (size_t)(m0 + arow) * K + (it + 2) * 32 + ac0, 0, 1);
        __builtin_prefetch(Bm + (size_t)((it + 2) * 32 + bkk) * N + n0 + bnc0, 0, 1);
      }
    }
    // compute on current buffer: 2 A-frags reused across 4 B-frags -> 8 WMMA
    v16bf af[2];
#pragma unroll
    for (int mi = 0; mi < 2; ++mi)
      af[mi] = frag_a_rm(&lds_a[cur][0][0], 32, m_off + mi * 16, 0, lane);
#pragma unroll
    for (int nj = 0; nj < 4; ++nj) {
      v16bf bf_ = frag_b_tr(&lds_bt[cur][0][0], 32, n_off + nj * 16, 0, lane);
#pragma unroll
      for (int mi = 0; mi < 2; ++mi)
        acc[mi][nj] = wmma_bf16_(af[mi], bf_, acc[mi][nj]);
    }
    __syncthreads();
    if (it + 1 < nsteps) {
#pragma unroll
      for (int j = 0; j < 16; ++j) lds_a[nxt][arow][ac0 + j] = ra[j];
#pragma unroll
      for (int j = 0; j < 16; ++j) lds_bt[nxt][bnc0 + j][bkk] = rb[j];
      __syncthreads();
    }
  }

  int kg = lane >> 4, nl = lane & 15;
#pragma unroll
  for (int mi = 0; mi < 2; ++mi)
#pragma unroll
    for (int nj = 0; nj < 4; ++nj)
#pragma unroll
      for (int i = 0; i < 8; ++i) {
        int mg = m0 + m_off + mi * 16 + 8 * kg + i;
        int ng = n0 + n_off + nj * 16 + nl;
        float v = acc[mi][nj][i];
        int b = mg >> 12, l = mg & (LL - 1);
        if (mode == 0) {
          if (ng < HH * DHD) {
            int h = ng >> 7, d = ng & 127;
            out_bf[(((size_t)(b * HH + h)) * LL + l) * DHD + d] = (__bf16)v;
          } else {
            int j2 = ng - HH * DHD;
            int h = j2 >> 7, d = j2 & 127;
            out_f[(((size_t)(b * HH + h)) * LL + l) * DHD + d] = v;
          }
        } else if (mode == 1) {
          int h = ng >> 7, d = ng & 127;
          out_bf[(((size_t)(b * HH + h)) * LL + l) * DHD + d] = (__bf16)v;
        } else {
          int lo = l + (CC - 1);
          if (lo < LL) out_f[((size_t)b * LL + lo) * DIMD + ng] = v;
        }
      }
}

// ============================================================
// 6) per-chunk 2-layer-MLP gradient: writes surprise = -grad(w0), -grad(w1)
//    One workgroup (8 waves) per chunk; all operands resident in LDS (160 KB).
// ============================================================
__global__ __launch_bounds__(256) void chunk_grad_k(const __bf16* __restrict__ keys,
                                                    const float* __restrict__ vals,
                                                    const float* __restrict__ lr_g,
                                                    const __bf16* __restrict__ w0b,
                                                    const __bf16* __restrict__ w1b,
                                                    float* __restrict__ s0_out,
                                                    float* __restrict__ s1_out) {
  extern __shared__ char smem[];
  __bf16* k_s  = (__bf16*)smem;           // 64x128
  __bf16* w0_s = k_s  + 64 * 128;         // 128x128
  __bf16* w1_s = w0_s + 128 * 128;        // 128x128
  __bf16* a_s  = w1_s + 128 * 128;        // 64x128
  __bf16* d_s  = a_s  + 64 * 128;         // 64x128
  __bf16* dh_s = d_s  + 64 * 128;         // 64x128
  float*  h_s  = (float*)(dh_s + 64 * 128); // 64x128 f32

  int cid = blockIdx.x;                   // bh*N + n
  int bh = cid >> 6, nch = cid & (NN - 1);
  int tid = threadIdx.x, lane = tid & 31, wv = tid >> 5;
  int kg = lane >> 4, nl = lane & 15;

  {  // stage in k, w0, w1
    int row = tid >> 2, c0 = (tid & 3) * 32;
    const __bf16* g = keys + ((size_t)bh * LL + nch * CC) * DHD;
#pragma unroll
    for (int j = 0; j < 32; ++j) k_s[row * 128 + c0 + j] = g[(size_t)row * 128 + c0 + j];
    int base = tid * 64;
#pragma unroll
    for (int j = 0; j < 64; ++j) { w0_s[base + j] = w0b[base + j]; w1_s[base + j] = w1b[base + j]; }
  }
  __syncthreads();

  // ---- S1: h = k @ w0 ; a = silu(h) ----
  {
    int mt = (wv >> 1) * 16, nt0 = (wv & 1) * 4;
    v8f acc[4] = {};
    for (int k0 = 0; k0 < 128; k0 += 32) {
      v16bf af = frag_a_rm(k_s, 128, mt, k0, lane);
#pragma unroll
      for (int j = 0; j < 4; ++j)
        acc[j] = wmma_bf16_(af, frag_b_rm(w0_s, 128, (nt0 + j) * 16, k0, lane), acc[j]);
    }
#pragma unroll
    for (int j = 0; j < 4; ++j)
#pragma unroll
      for (int i = 0; i < 8; ++i) {
        int m = mt + 8 * kg + i, n = (nt0 + j) * 16 + nl;
        float hv = acc[j][i];
        h_s[m * 128 + n] = hv;
        a_s[m * 128 + n] = (__bf16)siluf_(hv);
      }
  }
  __syncthreads();

  // ---- S2: pred = a @ w1 ; d = (2/DH)*lr*(pred - v) ----
  {
    int mt = (wv >> 1) * 16, nt0 = (wv & 1) * 4;
    v8f acc[4] = {};
    for (int k0 = 0; k0 < 128; k0 += 32) {
      v16bf af = frag_a_rm(a_s, 128, mt, k0, lane);
#pragma unroll
      for (int j = 0; j < 4; ++j)
        acc[j] = wmma_bf16_(af, frag_b_rm(w1_s, 128, (nt0 + j) * 16, k0, lane), acc[j]);
    }
#pragma unroll
    for (int j = 0; j < 4; ++j)
#pragma unroll
      for (int i = 0; i < 8; ++i) {
        int t = mt + 8 * kg + i, n = (nt0 + j) * 16 + nl;
        int tok = nch * CC + t;
        float lr = lr_g[(size_t)bh * LL + tok];
        float vv = vals[((size_t)bh * LL + tok) * DHD + n];
        d_s[t * 128 + n] = (__bf16)((lr * (2.0f / DHD)) * (acc[j][i] - vv));
      }
  }
  __syncthreads();

  // ---- S3: da = d @ w1ᵀ ; dh = da * silu'(h) ----
  {
    int mt = (wv >> 1) * 16, nt0 = (wv & 1) * 4;
    v8f acc[4] = {};
    for (int k0 = 0; k0 < 128; k0 += 32) {
      v16bf af = frag_a_rm(d_s, 128, mt, k0, lane);
#pragma unroll
      for (int j = 0; j < 4; ++j)  // B[k][n] = w1[n][k] -> contiguous view
        acc[j] = wmma_bf16_(af, frag_b_tr(w1_s, 128, (nt0 + j) * 16, k0, lane), acc[j]);
    }
#pragma unroll
    for (int j = 0; j < 4; ++j)
#pragma unroll
      for (int i = 0; i < 8; ++i) {
        int m = mt + 8 * kg + i, n = (nt0 + j) * 16 + nl;
        float hv = h_s[m * 128 + n];
        float sg = sigmoidf_(hv);
        float sp = sg * (1.0f + hv * (1.0f - sg));   // silu'
        dh_s[m * 128 + n] = (__bf16)(acc[j][i] * sp);
      }
  }
  __syncthreads();

  // ---- S4: g1 = aᵀ @ d  -> write -g1 ----
  {
    int mt = wv * 16;
    v8f acc8[8] = {};
    for (int k0 = 0; k0 < 64; k0 += 32) {
      v16bf af = frag_a_tr(a_s, 128, mt, k0, lane);   // A[m][k] = a[k][m]
#pragma unroll
      for (int j = 0; j < 8; ++j)
        acc8[j] = wmma_bf16_(af, frag_b_rm(d_s, 128, j * 16, k0, lane), acc8[j]);
    }
#pragma unroll
    for (int j = 0; j < 8; ++j)
#pragma unroll
      for (int i = 0; i < 8; ++i) {
        int m = mt + 8 * kg + i, n = j * 16 + nl;
        s1_out[(size_t)cid * (DHD * DHD) + m * 128 + n] = -acc8[j][i];
      }
  }

  // ---- S5: g0 = kᵀ @ dh -> write -g0 ----
  {
    int mt = wv * 16;
    v8f acc8[8] = {};
    for (int k0 = 0; k0 < 64; k0 += 32) {
      v16bf af = frag_a_tr(k_s, 128, mt, k0, lane);
#pragma unroll
      for (int j = 0; j < 8; ++j)
        acc8[j] = wmma_bf16_(af, frag_b_rm(dh_s, 128, j * 16, k0, lane), acc8[j]);
    }
#pragma unroll
    for (int j = 0; j < 8; ++j)
#pragma unroll
      for (int i = 0; i < 8; ++i) {
        int m = mt + 8 * kg + i, n = j * 16 + nl;
        s0_out[(size_t)cid * (DHD * DHD) + m * 128 + n] = -acc8[j][i];
      }
  }
}

// ============================================================
// 7) double linear scan over chunks + add base weights, emit bf16 w0_t/w1_t
// ============================================================
__global__ __launch_bounds__(256) void scan_k(const float* __restrict__ s0,
                                              const float* __restrict__ s1,
                                              const float* __restrict__ adam,
                                              const float* __restrict__ dec,
                                              const float* __restrict__ w0f,
                                              const float* __restrict__ w1f,
                                              __bf16* __restrict__ w0t,
                                              __bf16* __restrict__ w1t) {
  int idx = blockIdx.x * 256 + threadIdx.x;   // < BH * DH*DH
  int bh = idx >> 14, e = idx & (DHD * DHD - 1);
  float m0 = 0.f, u0 = 0.f, m1 = 0.f, u1 = 0.f;
  float w0v = w0f[e], w1v = w1f[e];
  for (int n = 0; n < NN; ++n) {
    float am = adam[(size_t)bh * NN + n];
    float od = 1.0f - dec[(size_t)bh * NN + n];
    size_t off = ((size_t)bh * NN + n) * (DHD * DHD) + e;
    m0 = am * m0 + s0[off]; u0 = od * u0 + m0;
    m1 = am * m1 + s1[off]; u1 = od * u1 + m1;
    w0t[off] = (__bf16)(w0v + u0);
    w1t[off] = (__bf16)(w1v + u1);
  }
}

// ============================================================
// 8) per-chunk retrieval MLP + MultiheadRMSNorm*(gamma+1)*gate (128 KB LDS)
// ============================================================
__global__ __launch_bounds__(256) void retrieve_k(const __bf16* __restrict__ qg,
                                                  const __bf16* __restrict__ w0t,
                                                  const __bf16* __restrict__ w1t,
                                                  const float* __restrict__ gamma,
                                                  const float* __restrict__ gate_g,
                                                  __bf16* __restrict__ outc) {
  extern __shared__ char smem[];
  __bf16* q_s  = (__bf16*)smem;            // 64x128
  __bf16* w0_s = q_s  + 64 * 128;          // 128x128
  __bf16* w1_s = w0_s + 128 * 128;         // 128x128
  __bf16* a_s  = w1_s + 128 * 128;         // 64x128
  float*  o_s  = (float*)(a_s + 64 * 128); // 64x128 f32

  int cid = blockIdx.x;
  int bh = cid >> 6, nch = cid & (NN - 1);
  int b = bh >> 2, h = bh & 3;
  int tid = threadIdx.x, lane = tid & 31, wv = tid >> 5;
  int kg = lane >> 4, nl = lane & 15;

  {
    int row = tid >> 2, c0 = (tid & 3) * 32;
    const __bf16* g = qg + ((size_t)bh * LL + nch * CC) * DHD;
#pragma unroll
    for (int j = 0; j < 32; ++j) q_s[row * 128 + c0 + j] = g[(size_t)row * 128 + c0 + j];
    int base = tid * 64;
    const __bf16* g0 = w0t + (size_t)cid * (DHD * DHD);
    const __bf16* g1 = w1t + (size_t)cid * (DHD * DHD);
#pragma unroll
    for (int j = 0; j < 64; ++j) { w0_s[base + j] = g0[base + j]; w1_s[base + j] = g1[base + j]; }
  }
  __syncthreads();

  // S1: a = silu(q @ w0_t)
  {
    int mt = (wv >> 1) * 16, nt0 = (wv & 1) * 4;
    v8f acc[4] = {};
    for (int k0 = 0; k0 < 128; k0 += 32) {
      v16bf af = frag_a_rm(q_s, 128, mt, k0, lane);
#pragma unroll
      for (int j = 0; j < 4; ++j)
        acc[j] = wmma_bf16_(af, frag_b_rm(w0_s, 128, (nt0 + j) * 16, k0, lane), acc[j]);
    }
#pragma unroll
    for (int j = 0; j < 4; ++j)
#pragma unroll
      for (int i = 0; i < 8; ++i) {
        int m = mt + 8 * kg + i, n = (nt0 + j) * 16 + nl;
        a_s[m * 128 + n] = (__bf16)siluf_(acc[j][i]);
      }
  }
  __syncthreads();

  // S2: o = a @ w1_t
  {
    int mt = (wv >> 1) * 16, nt0 = (wv & 1) * 4;
    v8f acc[4] = {};
    for (int k0 = 0; k0 < 128; k0 += 32) {
      v16bf af = frag_a_rm(a_s, 128, mt, k0, lane);
#pragma unroll
      for (int j = 0; j < 4; ++j)
        acc[j] = wmma_bf16_(af, frag_b_rm(w1_s, 128, (nt0 + j) * 16, k0, lane), acc[j]);
    }
#pragma unroll
    for (int j = 0; j < 4; ++j)
#pragma unroll
      for (int i = 0; i < 8; ++i) {
        int m = mt + 8 * kg + i, n = (nt0 + j) * 16 + nl;
        o_s[m * 128 + n] = acc[j][i];
      }
  }
  __syncthreads();

  // rowwise rmsnorm over DH, *(gamma+1), *gate, pack to (B, L, H*DH)
  if (tid < CC) {
    float ss = 0.0f;
#pragma unroll
    for (int d = 0; d < DHD; ++d) { float x = o_s[tid * 128 + d]; ss += x * x; }
    float rinv = rsqrtf(ss * (1.0f / DHD) + EPSF);
    int l = nch * CC + tid;
    float gt = gate_g[(size_t)bh * LL + l];
    size_t ob = ((size_t)b * LL + l) * (HH * DHD) + h * DHD;
#pragma unroll
    for (int d = 0; d < DHD; ++d)
      outc[ob + d] = (__bf16)(o_s[tid * 128 + d] * rinv * (gamma[h * DHD + d] + 1.0f) * gt);
  }
}

// ============================================================
// 9) fill d_out[:, 0:63, :] with empty_embed
// ============================================================
__global__ void fill_empty_k(const float* __restrict__ emb, float* __restrict__ out) {
  int i = blockIdx.x * 256 + threadIdx.x;           // < B*63*DIM = 64512
  if (i < BB * (CC - 1) * DIMD) {
    int b = i / ((CC - 1) * DIMD);
    int r = i % ((CC - 1) * DIMD);
    int l = r >> 9, d = r & (DIMD - 1);
    out[((size_t)b * LL + l) * DIMD + d] = emb[d];
  }
}

// ============================================================
// host launcher
// ============================================================
extern "C" void kernel_launch(void* const* d_in, const int* in_sizes, int n_in,
                              void* d_out, int out_size, void* d_ws, size_t ws_size,
                              hipStream_t stream) {
  const float* seq     = (const float*)d_in[0];
  const float* ssc     = (const float*)d_in[1];
  const float* rsc     = (const float*)d_in[2];
  const float* w_q     = (const float*)d_in[3];
  const float* w_kv    = (const float*)d_in[4];
  const float* w_ada   = (const float*)d_in[5];
  const float* w_mom   = (const float*)d_in[6];
  const float* w_dec   = (const float*)d_in[7];
  const float* w0      = (const float*)d_in[8];
  const float* w1      = (const float*)d_in[9];
  const float* gamma   = (const float*)d_in[10];
  const float* w_gate  = (const float*)d_in[11];
  const float* w_comb  = (const float*)d_in[12];
  const float* emb     = (const float*)d_in[13];
  float* out = (float*)d_out;

  char* base = (char*)d_ws;
  size_t off = 0;
  auto alloc = [&](size_t bytes) { char* p = base + off; off = (off + bytes + 255) & ~(size_t)255; return p; };

  const size_t MD = (size_t)MM * DIMD;           // 4,194,304
  __bf16* s_bf    = (__bf16*)alloc(MD * 2);
  __bf16* rsh_bf  = (__bf16*)alloc(MD * 2);
  __bf16* keys_bf = (__bf16*)alloc(MD * 2);
  __bf16* q_bf    = (__bf16*)alloc(MD * 2);
  float*  vals_f  = (float*)alloc(MD * 4);
  float*  lr_f    = (float*)alloc((size_t)BHH * LL * 4);
  float*  gate_f  = (float*)alloc((size_t)BHH * LL * 4);
  float*  adam_f  = (float*)alloc((size_t)BHH * NN * 4);
  float*  dec_f   = (float*)alloc((size_t)BHH * NN * 4);
  float*  g0_f    = (float*)alloc((size_t)BHH * NN * DHD * DHD * 4);
  float*  g1_f    = (float*)alloc((size_t)BHH * NN * DHD * DHD * 4);
  __bf16* w0t_bf  = (__bf16*)alloc((size_t)BHH * NN * DHD * DHD * 2);
  __bf16* w1t_bf  = (__bf16*)alloc((size_t)BHH * NN * DHD * DHD * 2);
  __bf16* outc_bf = (__bf16*)alloc(MD * 2);
  __bf16* wkv_bf  = (__bf16*)alloc((size_t)DIMD * 2 * HH * DHD * 2);
  __bf16* wq_bf   = (__bf16*)alloc((size_t)DIMD * HH * DHD * 2);
  __bf16* wcomb_bf= (__bf16*)alloc((size_t)HH * DHD * DIMD * 2);
  __bf16* w0_bf   = (__bf16*)alloc((size_t)DHD * DHD * 2);
  __bf16* w1_bf   = (__bf16*)alloc((size_t)DHD * DHD * 2);

  // 1) rmsnorms
  rmsnorm_k<<<MM, 256, 0, stream>>>(seq, ssc, rsc, s_bf, rsh_bf);

  // 2) weight conversions
  cvt_bf16_k<<<(DIMD * 2 * HH * DHD + 255) / 256, 256, 0, stream>>>(w_kv, wkv_bf, DIMD * 2 * HH * DHD);
  cvt_bf16_k<<<(DIMD * HH * DHD + 255) / 256, 256, 0, stream>>>(w_q, wq_bf, DIMD * HH * DHD);
  cvt_bf16_k<<<(HH * DHD * DIMD + 255) / 256, 256, 0, stream>>>(w_comb, wcomb_bf, HH * DHD * DIMD);
  cvt_bf16_k<<<(DHD * DHD + 255) / 256, 256, 0, stream>>>(w0, w0_bf, DHD * DHD);
  cvt_bf16_k<<<(DHD * DHD + 255) / 256, 256, 0, stream>>>(w1, w1_bf, DHD * DHD);

  // 3) tiny per-token / per-chunk projections
  tokproj_k<<<MM, 64, 0, stream>>>(s_bf, rsh_bf, w_ada, w_gate, lr_f, gate_f);
  chunkstat_k<<<BB * NN, 256, 0, stream>>>(s_bf, w_mom, w_dec, adam_f, dec_f);

  // 4) kv and q projections (WMMA, 128x128 tiles, double-buffered)
  gemm_proj_k<<<dim3(MM / 128, (2 * HH * DHD) / 128), 256, 0, stream>>>(
      s_bf, wkv_bf, 2 * HH * DHD, DIMD, 0, keys_bf, vals_f);
  gemm_proj_k<<<dim3(MM / 128, (HH * DHD) / 128), 256, 0, stream>>>(
      rsh_bf, wq_bf, HH * DHD, DIMD, 1, q_bf, nullptr);

  // 5) per-chunk gradients (WMMA, 160 KB LDS per workgroup)
  chunk_grad_k<<<BHH * NN, 256, 160 * 1024, stream>>>(keys_bf, vals_f, lr_f, w0_bf, w1_bf, g0_f, g1_f);

  // 6) momentum/decay scans + base-weight add
  scan_k<<<(BHH * DHD * DHD) / 256, 256, 0, stream>>>(g0_f, g1_f, adam_f, dec_f, w0, w1, w0t_bf, w1t_bf);

  // 7) retrieval MLP + norm/gate (WMMA, 128 KB LDS)
  retrieve_k<<<BHH * NN, 256, 128 * 1024, stream>>>(q_bf, w0t_bf, w1t_bf, gamma, gate_f, outc_bf);

  // 8) combine heads with +63 shift (WMMA), then prefix fill
  gemm_proj_k<<<dim3(MM / 128, DIMD / 128), 256, 0, stream>>>(
      outc_bf, wcomb_bf, DIMD, HH * DHD, 2, nullptr, out);
  fill_empty_k<<<(BB * (CC - 1) * DIMD + 255) / 256, 256, 0, stream>>>(emb, out);
}
